// TtFalconAttention_35459249995875
// MI455X (gfx1250) — compile-verified, matrix operand order checked by reading, not run on hardware
//
#include <hip/hip_runtime.h>
#include <hip/hip_bf16.h>
#include <math.h>

// ---------------------------------------------------------------- constants
#define NHEADS 71
#define HD     64
#define SLEN   1024
#define HDIM   (NHEADS * HD)        // 4544
#define FDIM   ((NHEADS + 2) * HD)  // 4672
#define ATT_SCALE 0.125f            // 1/sqrt(64)

typedef unsigned int   u32;
typedef unsigned short u16;
typedef __attribute__((ext_vector_type(16))) __bf16 v16bf;
typedef __attribute__((ext_vector_type(8)))  float  v8f;
typedef __attribute__((ext_vector_type(4)))  u32    u32x4;
typedef __attribute__((ext_vector_type(4)))  int    i32x4;

union BfFrag { u32x4 q[2]; u32 u[8]; v16bf v; };

// ---- CDNA5 async global->LDS path (guarded: falls back to sync b128 copies)
#if defined(__has_builtin)
#  if __has_builtin(__builtin_amdgcn_global_load_async_to_lds_b128) && \
      __has_builtin(__builtin_amdgcn_s_wait_asynccnt)
#    define USE_ASYNC_LDS 1
#  endif
#endif
#ifndef USE_ASYNC_LDS
#  define USE_ASYNC_LDS 0
#endif

#if USE_ASYNC_LDS
// builtin signature (from hipcc diagnostic):
//   (int4 as(1)* gsrc, int4 as(3)* ldst, imm int offset, imm int cpol)
typedef __attribute__((address_space(1))) i32x4 gl_i32x4;
typedef __attribute__((address_space(3))) i32x4 lds_i32x4;
__device__ __forceinline__ gl_i32x4* to_global4(const void* p) {
    return (gl_i32x4*)(unsigned long long)p;
}
__device__ __forceinline__ lds_i32x4* to_lds4(const void* p) {
    return (lds_i32x4*)(u32)(unsigned long long)p;  // low 32 bits = LDS offset
}
#endif

__device__ __forceinline__ u16 f2bf(float f) {
    u32 u = __float_as_uint(f);
    u32 r = u + 0x7FFFu + ((u >> 16) & 1u);   // round-to-nearest-even
    return (u16)(r >> 16);
}

// ---------------------------------------------------------------- f32 -> bf16
__global__ void cvt_f32_bf16(const float* __restrict__ src,
                             u16* __restrict__ dst, int n) {
    int i = blockIdx.x * blockDim.x + threadIdx.x;
    if (i < n) dst[i] = f2bf(src[i]);
}

// ---------------------------------------------------------------- WMMA GEMM
// C[M,N] (f32) = A[M,K] (bf16 rowmajor) * B[K,N] (bf16 rowmajor)
// block tile 128(M) x 64(N), 8 waves -> each 32x32 (2x2 WMMA tiles),
// K-steps of 32, double-buffered LDS; A staged async-to-LDS when available,
// B staged with a register transpose into BsT[col][k] so B fragments are
// contiguous ds_load_b128s.
__device__ __forceinline__ void gemm_stage(const u16* __restrict__ A,
                                           const u16* __restrict__ B,
                                           int N, int K, int bm, int bn,
                                           int kb, int tid,
                                           u16* As, u16* BsT) {
    // ---- A tile 128x32 (8 KB) : 512 x 16B chunks, 2 per thread
#if USE_ASYNC_LDS
#pragma unroll
    for (int i = tid; i < 512; i += 256) {
        int row = i >> 2, qq = i & 3;
        __builtin_amdgcn_global_load_async_to_lds_b128(
            to_global4(A + (size_t)(bm + row) * K + kb + qq * 8),
            to_lds4(As + row * 32 + qq * 8), 0, 0);
    }
#else
#pragma unroll
    for (int i = tid; i < 512; i += 256) {
        int row = i >> 2, qq = i & 3;
        *(u32x4*)(As + row * 32 + qq * 8) =
            *(const u32x4*)(A + (size_t)(bm + row) * K + kb + qq * 8);
    }
#endif
    // ---- B tile 32x64 -> transposed BsT[64][32]: register transpose
#pragma unroll
    for (int w = tid; w < 512; w += 256) {
        int c2 = w & 31;        // column pair 0..31
        int rp = w >> 5;        // row pair    0..15
        const u16* g = B + (size_t)(kb + 2 * rp) * N + bn + 2 * c2;
        u32 g0 = *(const u32*)g;          // B[r0][c], B[r0][c+1]
        u32 g1 = *(const u32*)(g + N);    // B[r1][c], B[r1][c+1]
        u32 lo = (g0 & 0xFFFFu) | (g1 << 16);
        u32 hi = (g0 >> 16) | (g1 & 0xFFFF0000u);
        u32* bt = (u32*)BsT;
        bt[(2 * c2) * 16 + rp]     = lo;
        bt[(2 * c2 + 1) * 16 + rp] = hi;
    }
    // ---- prefetch next-next tiles into GL2 (global_prefetch_b8)
    if (kb + 64 < K) {
        __builtin_prefetch(A + (size_t)(bm + (tid >> 1)) * K + kb + 64 +
                               (tid & 1) * 16, 0, 1);
        __builtin_prefetch(B + (size_t)(kb + 64 + (tid >> 3)) * N + bn +
                               (tid & 7) * 8, 0, 1);
    }
}

__launch_bounds__(256)
__global__ void gemm_bf16_wmma(const u16* __restrict__ A,
                               const u16* __restrict__ B,
                               float* __restrict__ C,
                               int M, int N, int K) {
    __shared__ __align__(16) u16 As[2][128 * 32];   // 2 x 8 KB
    __shared__ __align__(16) u16 BsT[2][64 * 32];   // 2 x 4 KB, [col][k]

    const int tid  = threadIdx.x;
    const int lane = tid & 31;
    const int wave = tid >> 5;
    const int wm   = wave & 3;
    const int wn   = wave >> 2;
    const int bm   = blockIdx.y * 128;
    const int bn   = blockIdx.x * 64;

    const int m  = lane & 15;
    const int ks = (lane & 16) ? 8 : 0;   // K-subgroup select (ISA 7.12.2)

    v8f acc[2][2];
#pragma unroll
    for (int i = 0; i < 2; ++i)
#pragma unroll
        for (int j = 0; j < 2; ++j)
#pragma unroll
            for (int e = 0; e < 8; ++e) acc[i][j][e] = 0.0f;

    const int nkt = K / 32;
    gemm_stage(A, B, N, K, bm, bn, 0, tid, As[0], BsT[0]);

    for (int kt = 0; kt < nkt; ++kt) {
        const int  cur     = kt & 1;
        const bool hasnext = (kt + 1) < nkt;
        if (hasnext)
            gemm_stage(A, B, N, K, bm, bn, (kt + 1) * 32, tid,
                       As[cur ^ 1], BsT[cur ^ 1]);
#if USE_ASYNC_LDS
        if (hasnext) __builtin_amdgcn_s_wait_asynccnt(2);
        else         __builtin_amdgcn_s_wait_asynccnt(0);
#endif
        __syncthreads();

        BfFrag af[2], bfr[2];
#pragma unroll
        for (int t = 0; t < 2; ++t) {
            const u16* ar = As[cur] + (wm * 32 + t * 16 + m) * 32;
            af[t].q[0] = *(const u32x4*)(ar + ks);
            af[t].q[1] = *(const u32x4*)(ar + 16 + ks);
            const u16* bc = BsT[cur] + (wn * 32 + t * 16 + m) * 32;
            bfr[t].q[0] = *(const u32x4*)(bc + ks);
            bfr[t].q[1] = *(const u32x4*)(bc + 16 + ks);
        }
#pragma unroll
        for (int i = 0; i < 2; ++i)
#pragma unroll
            for (int j = 0; j < 2; ++j)
                acc[i][j] = __builtin_amdgcn_wmma_f32_16x16x32_bf16(
                    false, af[i].v, false, bfr[j].v, (short)0, acc[i][j],
                    false, false);
        __syncthreads();
    }

    const int n0    = bn + wn * 32 + (lane & 15);
    const int mbase = bm + wm * 32 + ((lane >> 4) << 3);
#pragma unroll
    for (int i = 0; i < 2; ++i)
#pragma unroll
        for (int j = 0; j < 2; ++j)
#pragma unroll
            for (int v = 0; v < 8; ++v)
                C[(size_t)(mbase + i * 16 + v) * N + n0 + j * 16] =
                    acc[i][j][v];
}

// ---------------------------------------------------------------- RoPE+split
// qkv f32 [S,F] -> q_b bf16 [NH,S,HD] (roped), k_b bf16 [S,HD] (roped),
// v_t bf16 [HD,S] (transposed, so attention B-fragments are contiguous)
__global__ void rope_split_kernel(const float* __restrict__ qkv,
                                  u16* __restrict__ q_b,
                                  u16* __restrict__ k_b,
                                  u16* __restrict__ v_t) {
    const int s = blockIdx.x;
    const int y = blockIdx.y;          // head id, or NHEADS for k/v
    const int d = threadIdx.x;         // 0..63

    const float invf = __powf(10000.0f, -2.0f * (float)(d & 31) / (float)HD);
    const float ang  = (float)s * invf;
    const float c    = __cosf(ang);
    const float si   = __sinf(ang);
    const int   dro  = (d < 32) ? d + 32 : d - 32;
    const float sgn  = (d < 32) ? -1.0f : 1.0f;

    if (y < NHEADS) {
        const float* row = qkv + (size_t)s * FDIM + y * HD;
        float val = row[d] * c + sgn * row[dro] * si;
        q_b[((size_t)y * SLEN + s) * HD + d] = f2bf(val);
    } else {
        const float* krow = qkv + (size_t)s * FDIM + HDIM;
        float kv = krow[d] * c + sgn * krow[dro] * si;
        k_b[(size_t)s * HD + d] = f2bf(kv);
        v_t[(size_t)d * SLEN + s] =
            f2bf(qkv[(size_t)s * FDIM + HDIM + HD + d]);
    }
}

// ---------------------------------------------------------------- attention
// One wave owns 16 queries of one head; flash-style online softmax; WMMA for
// both q*k^T and P*V.  ctx written bf16 merged [S, HDIM].
__launch_bounds__(256)
__global__ void attn_kernel(const u16* __restrict__ Qb,
                            const u16* __restrict__ Kb,
                            const u16* __restrict__ Vt,
                            u16* __restrict__ Ctx) {
    __shared__ __align__(16) u16 Pbuf[8][16 * 32];   // per-wave P tile, 8 KB

    const int lane = threadIdx.x & 31;
    const int wave = threadIdx.x >> 5;
    const int head = blockIdx.y;
    const int qb   = blockIdx.x * 128 + wave * 16;

    const int m       = lane & 15;
    const int ks      = (lane & 16) ? 8 : 0;
    const int halfsel = (lane >> 4) << 3;

    // --- q fragments (K = 64 -> two 16x32 A-fragments), loaded once
    BfFrag aq[2];
    {
        const u16* qrow = Qb + ((size_t)head * SLEN + qb + m) * HD;
#pragma unroll
        for (int kk = 0; kk < 2; ++kk) {
            aq[kk].q[0] = *(const u32x4*)(qrow + kk * 32 + ks);
            aq[kk].q[1] = *(const u32x4*)(qrow + kk * 32 + 16 + ks);
        }
    }

    v8f o[4];
#pragma unroll
    for (int n = 0; n < 4; ++n)
#pragma unroll
        for (int v = 0; v < 8; ++v) o[n][v] = 0.0f;
    float mrow[8], lrow[8];
#pragma unroll
    for (int v = 0; v < 8; ++v) { mrow[v] = -1e30f; lrow[v] = 0.0f; }

    const int nk = qb + 16;   // causal: keys [0, qb+16)
    for (int kc = 0; kc < nk; kc += 32) {
        // --- scores: S(16x32) = q(16x64) . k^T(64x32)
        v8f sc[2];
#pragma unroll
        for (int g = 0; g < 2; ++g)
#pragma unroll
            for (int v = 0; v < 8; ++v) sc[g][v] = 0.0f;
#pragma unroll
        for (int g = 0; g < 2; ++g) {
#pragma unroll
            for (int kk = 0; kk < 2; ++kk) {
                BfFrag bk;   // B = k^T: rows=d (contiguous in Kb), cols=key
                const u16* kr =
                    Kb + (size_t)(kc + g * 16 + m) * HD + kk * 32;
                bk.q[0] = *(const u32x4*)(kr + ks);
                bk.q[1] = *(const u32x4*)(kr + 16 + ks);
                sc[g] = __builtin_amdgcn_wmma_f32_16x16x32_bf16(
                    false, aq[kk].v, false, bk.v, (short)0, sc[g],
                    false, false);
            }
        }

        // --- scale + causal mask + online softmax update
#pragma unroll
        for (int v = 0; v < 8; ++v) {
            const int qidx = qb + halfsel + v;
            const int key0 = kc + (lane & 15);
            const int key1 = key0 + 16;
            float x0 = sc[0][v] * ATT_SCALE + ((key0 > qidx) ? -1e9f : 0.0f);
            float x1 = sc[1][v] * ATT_SCALE + ((key1 > qidx) ? -1e9f : 0.0f);

            float t = fmaxf(x0, x1);
#pragma unroll
            for (int w = 1; w < 16; w <<= 1)
                t = fmaxf(t, __shfl_xor(t, w, 32));
            const float mn   = fmaxf(mrow[v], t);
            const float corr = __expf(mrow[v] - mn);
            mrow[v] = mn;

            const float p0 = __expf(x0 - mn);
            const float p1 = __expf(x1 - mn);
            float rs = p0 + p1;
#pragma unroll
            for (int w = 1; w < 16; w <<= 1)
                rs += __shfl_xor(rs, w, 32);
            lrow[v] = lrow[v] * corr + rs;
#pragma unroll
            for (int n = 0; n < 4; ++n) o[n][v] *= corr;

            Pbuf[wave][(halfsel + v) * 32 + (lane & 15)]      = f2bf(p0);
            Pbuf[wave][(halfsel + v) * 32 + 16 + (lane & 15)] = f2bf(p1);
        }

        // --- repack P into A-fragment layout (per-wave LDS, wave-ordered)
        BfFrag pa;
        {
            const u16* pr = &Pbuf[wave][m * 32];
            pa.q[0] = *(const u32x4*)(pr + ks);
            pa.q[1] = *(const u32x4*)(pr + 16 + ks);
        }

        // --- ctx += P(16x32) . V(32x64); V transposed -> contiguous frags
#pragma unroll
        for (int n = 0; n < 4; ++n) {
            BfFrag bv;
            const u16* vc = Vt + (size_t)(n * 16 + m) * SLEN + kc;
            bv.q[0] = *(const u32x4*)(vc + ks);
            bv.q[1] = *(const u32x4*)(vc + 16 + ks);
            o[n] = __builtin_amdgcn_wmma_f32_16x16x32_bf16(
                false, pa.v, false, bv.v, (short)0, o[n], false, false);
        }
    }

    // --- normalize + store merged ctx (bf16)
#pragma unroll
    for (int n = 0; n < 4; ++n)
#pragma unroll
        for (int v = 0; v < 8; ++v) {
            float val = o[n][v] / lrow[v];
            int srow  = qb + halfsel + v;
            Ctx[(size_t)srow * HDIM + head * HD + n * 16 + (lane & 15)] =
                f2bf(val);
        }
}

// ---------------------------------------------------------------- launcher
extern "C" void kernel_launch(void* const* d_in, const int* in_sizes, int n_in,
                              void* d_out, int out_size, void* d_ws,
                              size_t ws_size, hipStream_t stream) {
    const float* hidden = (const float*)d_in[0];   // [1,1,S,HDIM]
    const float* wqkv   = (const float*)d_in[1];   // [HDIM, FDIM]
    const float* wdense = (const float*)d_in[2];   // [HDIM, HDIM]
    // d_in[3] = attention_mask: pure causal, applied analytically.

    char* ws = (char*)d_ws;
    size_t off = 0;
    auto alloc = [&](size_t bytes) -> char* {
        char* p = ws + off;
        off += (bytes + 255) & ~(size_t)255;
        return p;
    };

    u16*   xb   = (u16*)alloc((size_t)SLEN * HDIM * 2);
    u16*   wqb  = (u16*)alloc((size_t)HDIM * FDIM * 2);
    u16*   wdb  = (u16*)alloc((size_t)HDIM * HDIM * 2);
    float* qkvf = (float*)alloc((size_t)SLEN * FDIM * 4);
    u16*   qbm  = (u16*)alloc((size_t)NHEADS * SLEN * HD * 2);
    u16*   kbm  = (u16*)alloc((size_t)SLEN * HD * 2);
    u16*   vtm  = (u16*)alloc((size_t)HD * SLEN * 2);
    u16*   ctxb = (u16*)alloc((size_t)SLEN * HDIM * 2);

    {   // f32 -> bf16 conversions
        int n1 = SLEN * HDIM;
        cvt_f32_bf16<<<(n1 + 255) / 256, 256, 0, stream>>>(hidden, xb, n1);
        int n2 = HDIM * FDIM;
        cvt_f32_bf16<<<(n2 + 255) / 256, 256, 0, stream>>>(wqkv, wqb, n2);
        int n3 = HDIM * HDIM;
        cvt_f32_bf16<<<(n3 + 255) / 256, 256, 0, stream>>>(wdense, wdb, n3);
    }

    // QKV projection: [S,HDIM] x [HDIM,FDIM] -> [S,FDIM] f32
    gemm_bf16_wmma<<<dim3(FDIM / 64, SLEN / 128), 256, 0, stream>>>(
        xb, wqb, qkvf, SLEN, FDIM, HDIM);

    // RoPE + head split (q per head, shared k / transposed v)
    rope_split_kernel<<<dim3(SLEN, NHEADS + 1), 64, 0, stream>>>(
        qkvf, qbm, kbm, vtm);

    // multi-query flash attention -> merged ctx bf16
    attn_kernel<<<dim3(SLEN / 128, NHEADS), 256, 0, stream>>>(
        qbm, kbm, vtm, ctxb);

    // output projection: [S,HDIM] x [HDIM,HDIM] -> d_out f32
    gemm_bf16_wmma<<<dim3(HDIM / 64, SLEN / 128), 256, 0, stream>>>(
        ctxb, wdb, (float*)d_out, SLEN, HDIM, HDIM);
}
